// SingleHeadSelfAttention_13013750907435
// MI455X (gfx1250) — matrix operand order repaired and checked by reading, not compiled, not run on hardware
//
#include <hip/hip_runtime.h>
#include <hip/hip_bf16.h>
#include <math.h>

// ---------------------------------------------------------------------------
// Types for CDNA5 WMMA (wave32): A/B = 16 bf16 per lane, C/D = 8 f32 per lane.
// ---------------------------------------------------------------------------
typedef __attribute__((ext_vector_type(16))) __bf16 v16bf;
typedef __attribute__((ext_vector_type(8)))  __bf16 v8bf;
typedef __attribute__((ext_vector_type(8)))  float  v8f;
typedef __attribute__((__vector_size__(4 * sizeof(int)))) int v4i;

static __device__ __forceinline__ __bf16 f2bf(float f) {
  unsigned u = __builtin_bit_cast(unsigned, f);
  u += 0x7fffu + ((u >> 16) & 1u);          // round-to-nearest-even
  unsigned short h = (unsigned short)(u >> 16);
  return __builtin_bit_cast(__bf16, h);
}
static __device__ __forceinline__ float bf2f(__bf16 h) {
  unsigned short s = __builtin_bit_cast(unsigned short, h);
  unsigned u = ((unsigned)s) << 16;
  return __builtin_bit_cast(float, u);
}

// ---------------------------------------------------------------------------
// Async global->LDS copy path (CDNA5 ASYNCcnt machinery, ISA 15.18.3 op 95-98).
// Builtin prototype (from toolchain diagnostic): (v4i AS1*, v4i AS3*, Ii, Ii).
// ---------------------------------------------------------------------------
#if __has_builtin(__builtin_amdgcn_global_load_async_to_lds_b128)
#define HAVE_ASYNC_LDS 1
static __device__ __forceinline__ void async_cp16(const __bf16* g, __bf16* l) {
  // 16 bytes per lane, tracked with ASYNCcnt.
  __builtin_amdgcn_global_load_async_to_lds_b128(
      (__attribute__((address_space(1))) v4i*)g,
      (__attribute__((address_space(3))) v4i*)l, 0, 0);
}
#else
#define HAVE_ASYNC_LDS 0
#endif

static __device__ __forceinline__ void wait_async0() {
#if __has_builtin(__builtin_amdgcn_s_wait_asynccnt)
  __builtin_amdgcn_s_wait_asynccnt(0);
#elif HAVE_ASYNC_LDS
  asm volatile("s_wait_asynccnt 0x0" ::: "memory");
#endif
}

// ---------------------------------------------------------------------------
// Fragment loaders (ISA 7.12.2 layouts). Work for global or LDS source.
// A 16x32 bf16 (MxK), source row-major (M,K), ld = row stride:
//   lanes 0-15: M=lane,   K = {0..7, 16..23}
//   lanes16-31: M=lane-16,K = {8..15,24..31}
// ---------------------------------------------------------------------------
static __device__ __forceinline__ v16bf load_frag_a(const __bf16* base,
                                                    int ld, int row, int k0, int lane) {
  int m  = row + (lane & 15);
  int kk = k0 + ((lane & 16) >> 1);          // +8 for high half-wave
  const v8bf* p = (const v8bf*)(base + (size_t)m * ld + kk);
  v8bf lo = p[0];                            // K = kk+0..7
  v8bf hi = p[2];                            // K = kk+16..23
  v16bf f;
#pragma unroll
  for (int e = 0; e < 8; ++e) { f[e] = lo[e]; f[e + 8] = hi[e]; }
  return f;
}

// B 32x16 bf16 (KxN), supplied as its transpose row-major (N,K), ld = row stride:
//   lanes 0-15: N=lane,   K = k0+{0..7, 16..23}
//   lanes16-31: N=lane-16,K = k0+{8..15,24..31}
static __device__ __forceinline__ v16bf load_frag_b(const __bf16* bt,
                                                    int ld, int col, int k0, int lane) {
  int n  = col + (lane & 15);
  int kk = k0 + ((lane & 16) >> 1);          // +8 for high half-wave
  const v8bf* p = (const v8bf*)(bt + (size_t)n * ld + kk);
  v8bf lo = p[0];
  v8bf hi = p[2];
  v16bf f;
#pragma unroll
  for (int e = 0; e < 8; ++e) { f[e] = lo[e]; f[e + 8] = hi[e]; }
  return f;
}

// ---------------------------------------------------------------------------
// Generic batched bf16 WMMA GEMM:  C[b] = scale * (A[b] @ Bt[b]^T) + bias
//   A  : (M,K) row-major bf16, row stride lda, batch stride sA (elements)
//   Bt : (N,K) row-major bf16 (i.e. B transposed), row stride ldbt, batch sB
//   C  : (M,N) row-major, stride ldc, batch sC; f32 or bf16 per out_is_bf16
// Block = 256 threads = 8 waves (2x4). Wave tile = 32x64 -> 8 WMMAs / K-step.
// Block tile 64x256 staged in LDS per 32-wide K-step via double-buffered
// async global->LDS copies: issue next tile, compute current from LDS,
// one s_wait_asynccnt + one barrier per step. LDS rows padded to 40 elems
// (80 B) so 16 consecutive rows hit 16 distinct bank groups (conflict-free
// ds_load_b128). All dims are exact multiples of tiles (no edges; EXEC all-1s
// for WMMA; no early returns so barriers stay converged).
// ---------------------------------------------------------------------------
#define LDSP 40

__global__ __launch_bounds__(256) void gemm_bf16_wmma(
    const __bf16* __restrict__ A, const __bf16* __restrict__ Bt,
    const float* __restrict__ bias, void* __restrict__ Cout,
    int M, int N, int K, int lda, int ldbt, int ldc,
    size_t sA, size_t sB, size_t sC, float scale, int out_is_bf16) {
  const int t    = threadIdx.x;
  const int lane = t & 31;
  const int w    = t >> 5;
  const int wm   = (w >> 2) * 32;            // wave M origin inside block tile
  const int wn   = (w & 3) * 64;             // wave N origin inside block tile
  const int rowB = blockIdx.y * 64;
  const int colB = blockIdx.x * 256;

  const __bf16* Ab = A  + (size_t)blockIdx.z * sA;
  const __bf16* Bb = Bt + (size_t)blockIdx.z * sB;

  v8f acc[2][4] = {};

#if HAVE_ASYNC_LDS
  __shared__ __attribute__((aligned(16))) __bf16 As[2][64][LDSP];
  __shared__ __attribute__((aligned(16))) __bf16 Bs[2][256][LDSP];

  // Per-thread fill assignment: A tile 64x32 = 256 chunks of 16B -> 1/thread;
  // B tile 256x32 = 1024 chunks -> 4/thread (row t, 4 segments).
  const int ar = t >> 2, aseg = (t & 3) * 8;
  const __bf16* agp = Ab + (size_t)(rowB + ar) * lda + aseg;
  const __bf16* bgp = Bb + (size_t)(colB + t) * ldbt;

  auto issue_tile = [&](int buf, int k0) {
    async_cp16(agp + k0, &As[buf][ar][aseg]);
#pragma unroll
    for (int s2 = 0; s2 < 4; ++s2)
      async_cp16(bgp + k0 + s2 * 8, &Bs[buf][t][s2 * 8]);
  };

  issue_tile(0, 0);
  const int nsteps = K >> 5;
  for (int s = 0; s < nsteps; ++s) {
    const int buf = s & 1;
    wait_async0();            // our 5 async fills for buf are done
    __syncthreads();          // tile visible to the whole workgroup
    if (s + 1 < nsteps) issue_tile(buf ^ 1, (s + 1) << 5);  // overlap with math

    const __bf16* Asb = &As[buf][0][0];
    const __bf16* Bsb = &Bs[buf][0][0];
    v16bf a[2], b[4];
    a[0] = load_frag_a(Asb, LDSP, wm,      0, lane);
    a[1] = load_frag_a(Asb, LDSP, wm + 16, 0, lane);
#pragma unroll
    for (int j = 0; j < 4; ++j)
      b[j] = load_frag_b(Bsb, LDSP, wn + j * 16, 0, lane);

#pragma unroll
    for (int i = 0; i < 2; ++i)
#pragma unroll
      for (int j = 0; j < 4; ++j)
        acc[i][j] = __builtin_amdgcn_wmma_f32_16x16x32_bf16(
            false, a[i], false, b[j], (short)0, acc[i][j], false, false);
  }
#else
  // Fallback: direct-from-global fragments (L2-resident working set).
  for (int k0 = 0; k0 < K; k0 += 32) {
    v16bf a[2], b[4];
    a[0] = load_frag_a(Ab, lda, rowB + wm,      k0, lane);
    a[1] = load_frag_a(Ab, lda, rowB + wm + 16, k0, lane);
#pragma unroll
    for (int j = 0; j < 4; ++j)
      b[j] = load_frag_b(Bb, ldbt, colB + wn + j * 16, k0, lane);
    if (k0 + 32 < K) {
      __builtin_prefetch(Ab + (size_t)(rowB + wm + (lane & 15)) * lda + k0 + 32, 0, 3);
      __builtin_prefetch(Bb + (size_t)(colB + wn + (lane & 15)) * ldbt + k0 + 32, 0, 3);
    }
#pragma unroll
    for (int i = 0; i < 2; ++i)
#pragma unroll
      for (int j = 0; j < 4; ++j)
        acc[i][j] = __builtin_amdgcn_wmma_f32_16x16x32_bf16(
            false, a[i], false, b[j], (short)0, acc[i][j], false, false);
  }
#endif

  // Epilogue: C/D layout -> lane 0-15: N=lane, M=r; lane 16-31: N=lane-16, M=r+8
  const int nbase = lane & 15;
  const int mofs  = (lane & 16) ? 8 : 0;
#pragma unroll
  for (int j = 0; j < 4; ++j) {
    const int n  = colB + wn + j * 16 + nbase;
    const float bv = bias ? bias[n] : 0.0f;
#pragma unroll
    for (int i = 0; i < 2; ++i) {
      const int mb = rowB + wm + i * 16 + mofs;
#pragma unroll
      for (int r = 0; r < 8; ++r) {
        const float v = acc[i][j][r] * scale + bv;
        const size_t idx = sC * blockIdx.z + (size_t)(mb + r) * ldc + n;
        if (out_is_bf16) ((__bf16*)Cout)[idx] = f2bf(v);
        else             ((float*)Cout)[idx]  = v;
      }
    }
  }
}

// ---------------------------------------------------------------------------
// float -> bf16 elementwise conversion
// ---------------------------------------------------------------------------
__global__ __launch_bounds__(256) void cvt_f32_bf16(const float* __restrict__ in,
                                                    __bf16* __restrict__ out, size_t n) {
  size_t i = (size_t)blockIdx.x * blockDim.x + threadIdx.x;
  if (i < n) out[i] = f2bf(in[i]);
}

// ---------------------------------------------------------------------------
// Row softmax in place on bf16 matrix, one block (256 threads) per row.
// ncols <= 2048 (= 8 values cached per thread).
// ---------------------------------------------------------------------------
__global__ __launch_bounds__(256) void softmax_rows(__bf16* __restrict__ mat, int ncols) {
  const int tid = threadIdx.x;
  __bf16* p = mat + (size_t)blockIdx.x * ncols;
  __shared__ float red[256];

  float xv[8];
  int   cnt  = 0;
  float lmax = -3.0e38f;
  for (int c = tid; c < ncols; c += 256) {
    float v = bf2f(p[c]);
    xv[cnt++] = v;
    lmax = fmaxf(lmax, v);
  }
  red[tid] = lmax;
  __syncthreads();
  for (int s = 128; s > 0; s >>= 1) {
    if (tid < s) red[tid] = fmaxf(red[tid], red[tid + s]);
    __syncthreads();
  }
  const float m = red[0];
  __syncthreads();

  float lsum = 0.0f;
  for (int j = 0; j < cnt; ++j) { xv[j] = expf(xv[j] - m); lsum += xv[j]; }
  red[tid] = lsum;
  __syncthreads();
  for (int s = 128; s > 0; s >>= 1) {
    if (tid < s) red[tid] += red[tid + s];
    __syncthreads();
  }
  const float inv = 1.0f / red[0];

  for (int j = 0; j < cnt; ++j) p[tid + j * 256] = f2bf(xv[j] * inv);
}

// ---------------------------------------------------------------------------
// Transpose V slice of the qkv buffer into vT[b][d][s] (bf16), LDS 32x32 tiles.
// ---------------------------------------------------------------------------
__global__ __launch_bounds__(256) void transpose_v(const __bf16* __restrict__ qkv,
                                                   __bf16* __restrict__ vT,
                                                   int S, int D, int ldq, int off) {
  __shared__ __bf16 tile[32][33];
  const int b  = blockIdx.z;
  const int d0 = blockIdx.x * 32;
  const int s0 = blockIdx.y * 32;
  const int tx = threadIdx.x & 31;
  const int ty = threadIdx.x >> 5;

  const __bf16* src = qkv + (size_t)b * S * ldq;
#pragma unroll
  for (int i = 0; i < 4; ++i) {
    const int sl = ty + i * 8;
    tile[sl][tx] = src[(size_t)(s0 + sl) * ldq + off + d0 + tx];
  }
  __syncthreads();
  __bf16* dst = vT + ((size_t)b * D + d0) * S + s0;
#pragma unroll
  for (int i = 0; i < 4; ++i) {
    const int dl = ty + i * 8;
    dst[(size_t)dl * S + tx] = tile[tx][dl];
  }
}

// ---------------------------------------------------------------------------
// Orchestration.
// ---------------------------------------------------------------------------
extern "C" void kernel_launch(void* const* d_in, const int* in_sizes, int n_in,
                              void* d_out, int out_size, void* d_ws, size_t ws_size,
                              hipStream_t stream) {
  (void)in_sizes; (void)n_in; (void)out_size; (void)ws_size;
  constexpr int B = 4, S = 2048, C = 2048, D = 512, TD = 3 * D;  // TD = 1536

  const float* x    = (const float*)d_in[0];   // (B,S,C)
  const float* Wqkv = (const float*)d_in[1];   // (3D,C)
  const float* bqkv = (const float*)d_in[2];   // (3D)
  const float* Wo   = (const float*)d_in[3];   // (C,D)
  const float* bo   = (const float*)d_in[4];   // (C)
  float* out = (float*)d_out;                  // (B,S,C)

  // Workspace layout (bf16); ~112 MB total, all chunks >=1MB so 16B-aligned.
  char* ws = (char*)d_ws;
  __bf16* xb   = (__bf16*)ws;  ws += (size_t)B * S * C  * 2;   // 33.5 MB
  __bf16* Wqb  = (__bf16*)ws;  ws += (size_t)TD * C     * 2;   //  6.3 MB
  __bf16* Wob  = (__bf16*)ws;  ws += (size_t)C * D      * 2;   //  2.1 MB
  __bf16* qkvb = (__bf16*)ws;  ws += (size_t)B * S * TD * 2;   // 25.2 MB
  __bf16* vT   = (__bf16*)ws;  ws += (size_t)B * D * S  * 2;   //  8.4 MB
  __bf16* scb  = (__bf16*)ws;  ws += (size_t)B * S * S  * 2;   // 33.5 MB (L2-resident)
  __bf16* ctxb = (__bf16*)ws;  ws += (size_t)B * S * D  * 2;   //  8.4 MB

  // 1) Downconvert activations + weights to bf16.
  {
    size_t n = (size_t)B * S * C;
    cvt_f32_bf16<<<dim3((unsigned)((n + 255) / 256)), dim3(256), 0, stream>>>(x, xb, n);
    n = (size_t)TD * C;
    cvt_f32_bf16<<<dim3((unsigned)((n + 255) / 256)), dim3(256), 0, stream>>>(Wqkv, Wqb, n);
    n = (size_t)C * D;
    cvt_f32_bf16<<<dim3((unsigned)((n + 255) / 256)), dim3(256), 0, stream>>>(Wo, Wob, n);
  }

  // 2) QKV projection: qkv = x @ Wqkv^T + bqkv.  M=B*S=8192, N=1536, K=2048.
  gemm_bf16_wmma<<<dim3(TD / 256, (B * S) / 64, 1), dim3(256), 0, stream>>>(
      xb, Wqb, bqkv, qkvb,
      B * S, TD, C, /*lda=*/C, /*ldbt=*/C, /*ldc=*/TD,
      0, 0, 0, 1.0f, /*bf16 out*/ 1);

  // 3) Transpose V (columns [1024,1536) of qkv) into vT[b][d][s].
  transpose_v<<<dim3(D / 32, S / 32, B), dim3(256), 0, stream>>>(
      qkvb, vT, S, D, TD, 2 * D);

  // 4) Scores: S[b] = (Q[b] @ K[b]^T) / sqrt(D).  Batched, M=N=2048, K=512.
  const float sc = 1.0f / sqrtf((float)D);
  gemm_bf16_wmma<<<dim3(S / 256, S / 64, B), dim3(256), 0, stream>>>(
      qkvb, qkvb + D, nullptr, scb,
      S, S, D, /*lda=*/TD, /*ldbt=*/TD, /*ldc=*/S,
      (size_t)S * TD, (size_t)S * TD, (size_t)S * S, sc, 1);

  // 5) Row softmax in place (B*S rows of length S).
  softmax_rows<<<dim3(B * S), dim3(256), 0, stream>>>(scb, S);

  // 6) Context: ctx[b] = P[b] @ V[b].  M=2048, N=512, K=2048; B^T == vT[b].
  gemm_bf16_wmma<<<dim3(D / 256, S / 64, B), dim3(256), 0, stream>>>(
      scb, vT, nullptr, ctxb,
      S, D, S, /*lda=*/S, /*ldbt=*/S, /*ldc=*/D,
      (size_t)S * S, (size_t)D * S, (size_t)S * D, 1.0f, 1);

  // 7) Output projection: out = ctx @ Wo^T + bo (f32 out). M=8192, N=2048, K=512.
  gemm_bf16_wmma<<<dim3(C / 256, (B * S) / 64, 1), dim3(256), 0, stream>>>(
      ctxb, Wob, bo, out,
      B * S, C, D, /*lda=*/D, /*ldbt=*/D, /*ldc=*/C,
      0, 0, 0, 1.0f, /*f32 out*/ 0);
}
